// Mwan_Imm_80255758893468
// MI455X (gfx1250) — matrix operand order repaired and checked by reading, not compiled, not run on hardware
//
#include <hip/hip_runtime.h>
#include <hip/hip_bf16.h>

// ---------------------------------------------------------------------------
// MultiWay attention (Concat / DotProduct / Minus / BiLinear) for MI455X.
// B=8, LQ=LP=128, H=256, HID=128.  All GEMM-shaped math runs on
// v_wmma_f32_16x16x32_bf16 (f32 accumulate).  Every WMMA operand fragment is
// fetched with two global_load_b128 per lane: B operands are pre-transposed
// once so both A and B fragments are contiguous 16-byte runs.
// ---------------------------------------------------------------------------

typedef __bf16 bf16_t;
typedef __attribute__((ext_vector_type(16))) __bf16 v16bf;
typedef __attribute__((ext_vector_type(8)))  float  v8f;

#define NB 8
#define LQ 128
#define LP 128
#define HH 256
#define HID 128

union frag_u {
  v16bf v;
  uint4 q[2];
};

// A (16x32 bf16, row-major src, tile origin pre-added, lda in elements).
// Lane r holds row r; K slots {half*8+0..7} and {16+half*8+0..7}: two
// contiguous 16-byte runs -> two b128 loads.
__device__ __forceinline__ v16bf load_a_frag(const bf16_t* __restrict__ A,
                                             int lda, int lane) {
  const int half = lane >> 4, r = lane & 15;
  const bf16_t* base = A + r * lda + half * 8;
  frag_u f;
  f.q[0] = *(const uint4*)(base);
  f.q[1] = *(const uint4*)(base + 16);
  return f.v;
}

// B fragment from a TRANSPOSED source BT[N x K] (row n of BT = column n of
// logical B[K x N]).  Lane n reads K = kh..kh+7 and 16+kh..16+kh+7.
__device__ __forceinline__ v16bf load_bt_frag(const bf16_t* __restrict__ BT,
                                              int ldb, int lane) {
  const int n = lane & 15, kh = (lane >> 4) * 8;
  const bf16_t* base = BT + n * ldb + kh;
  frag_u f;
  f.q[0] = *(const uint4*)(base);
  f.q[1] = *(const uint4*)(base + 16);
  return f.v;
}

// ---------------- elementwise conversions / transposes ---------------------

__global__ void f32_to_bf16_kernel(const float* __restrict__ src,
                                   bf16_t* __restrict__ dst, int n) {
  const int i = blockIdx.x * blockDim.x + threadIdx.x;
  if (i < n) dst[i] = (bf16_t)src[i];
}

// dst[N x K] bf16 = transpose(src[K x N] f32); batched over gridDim.z.
__global__ void transpose_f32_to_bf16(const float* __restrict__ src,
                                      bf16_t* __restrict__ dst, int K, int N) {
  const long long bz = blockIdx.z;
  src += bz * (long long)K * N;
  dst += bz * (long long)K * N;
  const int idx = blockIdx.x * blockDim.x + threadIdx.x;
  if (idx < K * N) {
    const int k = idx / N, n = idx % N;
    dst[n * K + k] = (bf16_t)src[idx];
  }
}

// hpb_bf[b,j,h] = bf16( hpb[b,j,h] * mask_hp[b,j] )
__global__ void mask_convert_hpb_kernel(const float* __restrict__ hpb,
                                        const float* __restrict__ mask_hp,
                                        bf16_t* __restrict__ hpb_bf, int n) {
  const int i = blockIdx.x * blockDim.x + threadIdx.x;
  if (i < n) hpb_bf[i] = (bf16_t)(hpb[i] * mask_hp[i >> 8]);  // /H=256
}

// ---------------- generic batched bf16 WMMA GEMM (B pre-transposed) --------
// C[M,N] (f32) = A[M,K] @ B[K,N] (+ bias[N]); BT is B transposed, N x K.
// One wave per 16x16 C tile.
__global__ void gemm_bf16_wmma(const bf16_t* __restrict__ A,
                               const bf16_t* __restrict__ BT,
                               const float* __restrict__ bias,
                               float* __restrict__ C,
                               int K, int lda, int ldb, int ldc,
                               long long sA, long long sB, long long sC) {
  const int lane = threadIdx.x;
  const int n0 = blockIdx.x * 16, m0 = blockIdx.y * 16, bz = blockIdx.z;
  A  += (long long)bz * sA;
  BT += (long long)bz * sB;
  C  += (long long)bz * sC;
  v8f acc = {0.f, 0.f, 0.f, 0.f, 0.f, 0.f, 0.f, 0.f};
  for (int k0 = 0; k0 < K; k0 += 32) {
    const v16bf a = load_a_frag(A + (long long)m0 * lda + k0, lda, lane);
    const v16bf b = load_bt_frag(BT + (long long)n0 * ldb + k0, ldb, lane);
    acc = __builtin_amdgcn_wmma_f32_16x16x32_bf16(false, a, false, b,
                                                  (short)0, acc, false, false);
  }
  const int half = lane >> 4, n = lane & 15;
  const float bn = bias ? bias[n0 + n] : 0.f;
#pragma unroll
  for (int v = 0; v < 8; ++v)
    C[(long long)(m0 + v + 8 * half) * ldc + n0 + n] = acc[v] + bn;
}

// ---------------- DotProductAttention scores (dominant term) ---------------
// For fixed (b, j): T = (hq ⊙ hp[j]) @ Wd, then s_d[b,i,j] = vd·tanh(T+bd).
// One wave handles a 16-row i-tile x full HID (8 accumulator tiles).
__global__ void score_d_wmma(const bf16_t* __restrict__ hq_bf,
                             const bf16_t* __restrict__ hp_bf,
                             const bf16_t* __restrict__ WdT,  // HID x H
                             const float* __restrict__ bd,
                             const float* __restrict__ vd,
                             float* __restrict__ s_all) {
  const int lane = threadIdx.x;
  const int it = blockIdx.x;  // LQ/16
  const int j  = blockIdx.y;  // LP
  const int b  = blockIdx.z;  // B
  const bf16_t* Aq = hq_bf + ((long long)b * LQ + it * 16) * HH;
  const bf16_t* Pj = hp_bf + ((long long)b * LP + j) * HH;
  const int half = lane >> 4, r = lane & 15, n = lane & 15;

  v8f acc[8];
#pragma unroll
  for (int t = 0; t < 8; ++t)
#pragma unroll
    for (int e = 0; e < 8; ++e) acc[t][e] = 0.f;

  for (int k0 = 0; k0 < HH; k0 += 32) {
    // A fragment scaled on the fly: hq[i,k] * hp[j,k]; all loads are b128.
    frag_u qa, pa;
    const bf16_t* qbase = Aq + r * HH + k0 + half * 8;
    const bf16_t* pbase = Pj + k0 + half * 8;  // lane-uniform per half
    qa.q[0] = *(const uint4*)(qbase);
    qa.q[1] = *(const uint4*)(qbase + 16);
    pa.q[0] = *(const uint4*)(pbase);
    pa.q[1] = *(const uint4*)(pbase + 16);
    v16bf a;
#pragma unroll
    for (int s = 0; s < 16; ++s)
      a[s] = (bf16_t)((float)qa.v[s] * (float)pa.v[s]);
#pragma unroll
    for (int nt = 0; nt < 8; ++nt) {
      const v16bf bf = load_bt_frag(WdT + (long long)nt * 16 * HH + k0,
                                    HH, lane);
      acc[nt] = __builtin_amdgcn_wmma_f32_16x16x32_bf16(
          false, a, false, bf, (short)0, acc[nt], false, false);
    }
  }

  // fused epilogue: part[v] = sum_h vd[h]*tanh(T[m,h]+bd[h]) (partial over N)
  float part[8];
#pragma unroll
  for (int v = 0; v < 8; ++v) part[v] = 0.f;
#pragma unroll
  for (int nt = 0; nt < 8; ++nt) {
    const int hc = nt * 16 + n;
    const float vdv = vd[hc], bdv = bd[hc];
#pragma unroll
    for (int v = 0; v < 8; ++v) part[v] += vdv * tanhf(acc[nt][v] + bdv);
  }
  // reduce across the 16 lanes that share row M (xor<16 stays in each half)
#pragma unroll
  for (int off = 8; off >= 1; off >>= 1)
#pragma unroll
    for (int v = 0; v < 8; ++v) part[v] += __shfl_xor(part[v], off, 32);

  if (n == 0) {
#pragma unroll
    for (int v = 0; v < 8; ++v) {
      const int i = it * 16 + v + 8 * half;
      s_all[((long long)(1 * NB + b) * LQ + i) * LP + j] = part[v];
    }
  }
}

// ---------------- Concat + Minus scores (pairwise tanh, VALU) --------------
__global__ void score_cm_kernel(const float* __restrict__ aqc,
                                const float* __restrict__ apc,
                                const float* __restrict__ vc,
                                const float* __restrict__ aqm,
                                const float* __restrict__ apm,
                                const float* __restrict__ vm,
                                float* __restrict__ s_all) {
  const int idx = blockIdx.x * blockDim.x + threadIdx.x;  // B*LQ*LP
  const int j = idx & 127, i = (idx >> 7) & 127, b = idx >> 14;
  const float4* qc = (const float4*)(aqc + ((long long)b * LQ + i) * HID);
  const float4* pc = (const float4*)(apc + ((long long)b * LP + j) * HID);
  const float4* qm = (const float4*)(aqm + ((long long)b * LQ + i) * HID);
  const float4* pm = (const float4*)(apm + ((long long)b * LP + j) * HID);
  const float4* vvc = (const float4*)vc;
  const float4* vvm = (const float4*)vm;
  float sc = 0.f, sm = 0.f;
#pragma unroll 4
  for (int h = 0; h < HID / 4; ++h) {
    const float4 a = qc[h], p = pc[h], v = vvc[h];
    sc += v.x * tanhf(a.x + p.x) + v.y * tanhf(a.y + p.y) +
          v.z * tanhf(a.z + p.z) + v.w * tanhf(a.w + p.w);
    const float4 am = qm[h], pv = pm[h], w = vvm[h];
    sm += w.x * tanhf(am.x - pv.x) + w.y * tanhf(am.y - pv.y) +
          w.z * tanhf(am.z - pv.z) + w.w * tanhf(am.w - pv.w);
  }
  s_all[((long long)(0 * NB + b) * LQ + i) * LP + j] = sc;
  s_all[((long long)(2 * NB + b) * LQ + i) * LP + j] = sm;
}

// ---------------- column softmax over i (axis=1), mask-aware ---------------
// Writes the attention matrix TRANSPOSED: aT[t,b, j, i] so the attend GEMM's
// A operand is a plain row-major fragment (contiguous b128 loads).
__global__ void softmax_col_kernel(const float* __restrict__ s_all,
                                   const float* __restrict__ mask_hq,
                                   const float* __restrict__ mask_hp,
                                   bf16_t* __restrict__ aT_bf) {
  const int i = threadIdx.x;                       // LQ
  const int j = blockIdx.x, b = blockIdx.y, t = blockIdx.z;
  const long long base = ((long long)(t * NB + b) * LQ) * LP;
  const float neg =
      (mask_hq[b * LQ + i] * mask_hp[b * LP + j] == 0.f) ? 10000.f : 0.f;
  const float x = s_all[base + (long long)i * LP + j] - neg;
  __shared__ float red[LQ];
  red[i] = x;
  __syncthreads();
  for (int s = LQ / 2; s > 0; s >>= 1) {
    if (i < s) red[i] = fmaxf(red[i], red[i + s]);
    __syncthreads();
  }
  const float mx = red[0];
  __syncthreads();
  const float e = __expf(x - mx);
  red[i] = e;
  __syncthreads();
  for (int s = LQ / 2; s > 0; s >>= 1) {
    if (i < s) red[i] += red[i + s];
    __syncthreads();
  }
  aT_bf[base + (long long)j * LQ + i] = (bf16_t)(e / red[0]);
}

// ---------------- attend: out[b,j,t*H+h] = sum_i a[t,b,i,j] * hq[b,i,h] ----
// A = aT (LP x LQ, row-major), BT = hqT (H x LQ, row-major) -> all b128.
__global__ void attend_wmma(const bf16_t* __restrict__ aT_bf,
                            const bf16_t* __restrict__ hqT_bf,
                            float* __restrict__ out) {
  const int lane = threadIdx.x;
  const int n0 = blockIdx.x * 16;   // h within H
  const int m0 = blockIdx.y * 16;   // j within LP
  const int tb = blockIdx.z;        // t*NB + b
  const int t = tb >> 3, b = tb & 7;
  const bf16_t* Aa = aT_bf + (long long)tb * LP * LQ;    // (j, i) row-major
  const bf16_t* Bh = hqT_bf + (long long)b * HH * LQ;    // (h, i) row-major
  v8f acc = {0.f, 0.f, 0.f, 0.f, 0.f, 0.f, 0.f, 0.f};
  for (int k0 = 0; k0 < LQ; k0 += 32) {
    const v16bf a = load_a_frag(Aa + (long long)m0 * LQ + k0, LQ, lane);
    const v16bf bf = load_bt_frag(Bh + (long long)n0 * LQ + k0, LQ, lane);
    acc = __builtin_amdgcn_wmma_f32_16x16x32_bf16(false, a, false, bf,
                                                  (short)0, acc, false, false);
  }
  const int half = lane >> 4, n = lane & 15;
#pragma unroll
  for (int v = 0; v < 8; ++v) {
    const int j = m0 + v + 8 * half;
    out[((long long)b * LP + j) * (4 * HH) + t * HH + n0 + n] = acc[v];
  }
}

// ---------------------------------------------------------------------------
extern "C" void kernel_launch(void* const* d_in, const int* in_sizes, int n_in,
                              void* d_out, int out_size, void* d_ws,
                              size_t ws_size, hipStream_t stream) {
  const float* hq      = (const float*)d_in[0];
  const float* hp      = (const float*)d_in[1];
  const float* mask_hq = (const float*)d_in[2];
  const float* mask_hp = (const float*)d_in[3];
  const float* Wc      = (const float*)d_in[4];
  const float* bc      = (const float*)d_in[5];
  const float* vc      = (const float*)d_in[6];
  const float* Wd      = (const float*)d_in[7];
  const float* bd      = (const float*)d_in[8];
  const float* vd      = (const float*)d_in[9];
  const float* Wm      = (const float*)d_in[10];
  const float* bm      = (const float*)d_in[11];
  const float* vm      = (const float*)d_in[12];
  const float* Wb      = (const float*)d_in[13];
  const float* bb      = (const float*)d_in[14];
  float* out = (float*)d_out;

  // ---- workspace bump allocator (~9 MB) ----
  char* ws = (char*)d_ws;
  size_t off = 0;
  auto alloc = [&](size_t bytes) {
    void* p = ws + off;
    off = (off + bytes + 255) & ~(size_t)255;
    return p;
  };
  const int nHQ = NB * LQ * HH;        // 262144
  bf16_t* hq_bf  = (bf16_t*)alloc(nHQ * 2);           // (b,i,h) row-major
  bf16_t* hp_bf  = (bf16_t*)alloc(nHQ * 2);           // (b,j,h) row-major
  bf16_t* hqT_bf = (bf16_t*)alloc(nHQ * 2);           // (b,h,i)
  bf16_t* WcT    = (bf16_t*)alloc(2 * HH * HID * 2);  // 2 x (HID x H)
  bf16_t* WdT    = (bf16_t*)alloc(HH * HID * 2);      // HID x H
  bf16_t* WmT    = (bf16_t*)alloc(HH * HID * 2);      // HID x H
  bf16_t* WbT    = (bf16_t*)alloc(HH * HH * 2);       // H x H
  float*  aq_c   = (float*)alloc(NB * LQ * HID * 4);
  float*  ap_c   = (float*)alloc(NB * LP * HID * 4);
  float*  aq_m   = (float*)alloc(NB * LQ * HID * 4);
  float*  ap_m   = (float*)alloc(NB * LP * HID * 4);
  float*  hpb_f  = (float*)alloc(NB * LP * HH * 4);
  bf16_t* hpb_bf = (bf16_t*)alloc(NB * LP * HH * 2);  // (b,j,h): already B^T
  float*  s_all  = (float*)alloc(4 * NB * LQ * LP * 4);
  bf16_t* aT_bf  = (bf16_t*)alloc(4 * NB * LQ * LP * 2);
  (void)ws_size; (void)n_in; (void)in_sizes; (void)out_size;

  // ---- 1) conversions and one-time transposes ----
  auto cvt = [&](const float* s, bf16_t* d, int n) {
    f32_to_bf16_kernel<<<(n + 255) / 256, 256, 0, stream>>>(s, d, n);
  };
  cvt(hq, hq_bf, nHQ);
  cvt(hp, hp_bf, nHQ);
  {
    dim3 blk(256);
    // hqT[b] = transpose(hq[b]): K=LQ rows, N=HH cols
    transpose_f32_to_bf16<<<dim3((LQ * HH + 255) / 256, 1, NB), blk, 0,
                            stream>>>(hq, hqT_bf, LQ, HH);
    transpose_f32_to_bf16<<<dim3((HH * HID + 255) / 256, 1, 1), blk, 0,
                            stream>>>(Wc, WcT, HH, HID);               // top
    transpose_f32_to_bf16<<<dim3((HH * HID + 255) / 256, 1, 1), blk, 0,
                            stream>>>(Wc + HH * HID, WcT + HH * HID, HH, HID);
    transpose_f32_to_bf16<<<dim3((HH * HID + 255) / 256, 1, 1), blk, 0,
                            stream>>>(Wd, WdT, HH, HID);
    transpose_f32_to_bf16<<<dim3((HH * HID + 255) / 256, 1, 1), blk, 0,
                            stream>>>(Wm, WmT, HH, HID);
    transpose_f32_to_bf16<<<dim3((HH * HH + 255) / 256, 1, 1), blk, 0,
                            stream>>>(Wb, WbT, HH, HH);
  }

  // ---- 2) projection GEMMs (M = B*L = 1024 rows, single batch) ----
  // aq_c = hq @ Wc[:H]            ; ap_c = hp @ Wc[H:] + bc
  // aq_m = hq @ Wm + bm           ; ap_m = hp @ Wm
  // hpb  = hp @ Wb + bb
  {
    dim3 blk(32);
    dim3 gP(HID / 16, NB * LQ / 16, 1);
    gemm_bf16_wmma<<<gP, blk, 0, stream>>>(hq_bf, WcT, nullptr, aq_c,
        HH, HH, HH, HID, 0, 0, 0);
    gemm_bf16_wmma<<<gP, blk, 0, stream>>>(hp_bf, WcT + HH * HID, bc, ap_c,
        HH, HH, HH, HID, 0, 0, 0);
    gemm_bf16_wmma<<<gP, blk, 0, stream>>>(hq_bf, WmT, bm, aq_m,
        HH, HH, HH, HID, 0, 0, 0);
    gemm_bf16_wmma<<<gP, blk, 0, stream>>>(hp_bf, WmT, nullptr, ap_m,
        HH, HH, HH, HID, 0, 0, 0);
    dim3 gB(HH / 16, NB * LP / 16, 1);
    gemm_bf16_wmma<<<gB, blk, 0, stream>>>(hp_bf, WbT, bb, hpb_f,
        HH, HH, HH, HH, 0, 0, 0);
  }

  // ---- 3) hpb * mask_hp -> bf16 (row-major (j,h) == B^T for s_b) ----
  {
    const int n = NB * LP * HH;
    mask_convert_hpb_kernel<<<(n + 255) / 256, 256, 0, stream>>>(
        hpb_f, mask_hp, hpb_bf, n);
  }

  // ---- 4) s_b = hq @ hpb^T (batched over B), into s_all slot 3 ----
  {
    dim3 blk(32), g(LP / 16, LQ / 16, NB);
    gemm_bf16_wmma<<<g, blk, 0, stream>>>(hq_bf, hpb_bf, nullptr,
        s_all + (long long)3 * NB * LQ * LP,
        HH, HH, HH, LP,
        (long long)LQ * HH, (long long)LP * HH, (long long)LQ * LP);
  }

  // ---- 5) s_c, s_m (pairwise tanh, slots 0 and 2) ----
  {
    const int n = NB * LQ * LP;
    score_cm_kernel<<<n / 256, 256, 0, stream>>>(aq_c, ap_c, vc,
                                                 aq_m, ap_m, vm, s_all);
  }

  // ---- 6) s_d (dominant WMMA term, slot 1) ----
  {
    dim3 blk(32), g(LQ / 16, LP, NB);
    score_d_wmma<<<g, blk, 0, stream>>>(hq_bf, hp_bf, WdT, bd, vd, s_all);
  }

  // ---- 7) masked softmax over i for all 4 scores (emits a^T in bf16) ----
  {
    dim3 blk(LQ), g(LP, NB, 4);
    softmax_col_kernel<<<g, blk, 0, stream>>>(s_all, mask_hq, mask_hp, aT_bf);
  }

  // ---- 8) attend: out[b, j, t*H + h] = a^T @ hq ----
  {
    dim3 blk(32), g(HH / 16, LP / 16, 4 * NB);
    attend_wmma<<<g, blk, 0, stream>>>(aT_bf, hqT_bf, out);
  }
}
